// FGNODE_64613488001249
// MI455X (gfx1250) — matrix operand order repaired and checked by reading, not compiled, not run on hardware
//
#include <hip/hip_runtime.h>
#include <hip/hip_bf16.h>

// ---------------------------------------------------------------------------
// CDNA5 (gfx1250) spectral neural-ODE pipeline.
// All heavy math runs through V_WMMA_F32_16X16X32_BF16; A-operand tiles are
// DMA'd into LDS by the Tensor Data Mover (tensor_load_to_lds + TENSORcnt).
// ---------------------------------------------------------------------------

typedef __attribute__((ext_vector_type(16))) __bf16 bf16x16;
typedef __attribute__((ext_vector_type(8)))  __bf16 bf16x8;
typedef __attribute__((ext_vector_type(8)))  float  f32x8;
typedef __attribute__((ext_vector_type(4)))  unsigned int u32x4;
typedef __attribute__((ext_vector_type(8)))  int    i32x8;
typedef __attribute__((ext_vector_type(4)))  int    i32x4;

#if __has_builtin(__builtin_amdgcn_tensor_load_to_lds) && __has_builtin(__builtin_amdgcn_s_wait_tensorcnt)
#define USE_TDM 1
#else
#define USE_TDM 0
#endif

#define BM 128
#define BN 128
#define BK 32
#define BSTR 56   // padded bf16 K-stride of transposed B tile (16B aligned, conflict-free)

#define NPTS 4096
#define IN_C 128
#define HID  512
#define OUT_C 64
#define NUM_LAYERS 8
#define NEG_SLOPE 0.01f

// ---------------------------------------------------------------------------

__device__ inline bf16x8 bzero8() {
    union { unsigned long long q[2]; bf16x8 v; } u;
    u.q[0] = 0ull; u.q[1] = 0ull;
    return u.v;
}

// A fragment (16x32 bf16) per ISA 7.12.2: lanes 0-15 -> rows, K chunks {0..7,16..23};
// lanes 16-31 -> same rows, K chunks {8..15,24..31}. Two ds_load_b128 per lane.
__device__ inline bf16x16 load_frag_a(const __bf16* as, int rowBase, int lane) {
    int r    = rowBase + (lane & 15);
    int half = (lane >> 4) * 8;
    union { bf16x16 v; bf16x8 h[2]; } u;
    u.h[0] = *(const bf16x8*)(as + r * BK + half);
    u.h[1] = *(const bf16x8*)(as + r * BK + half + 16);
    return u.v;
}

// B fragment (32x16 bf16): lane holds one column; lanes 0-15 K {0..7,16..23}, lanes 16-31 K {8..15,24..31},
// consecutive K packed in pairs per VGPR -> contiguous in our n-major LDS tile.
__device__ inline bf16x16 load_frag_b(const __bf16* bs, int colBase, int lane) {
    int n  = colBase + (lane & 15);
    int kb = (lane >> 4) * 8;
    union { bf16x16 v; bf16x8 h[2]; } u;
    u.h[0] = *(const bf16x8*)(bs + n * BSTR + kb);
    u.h[1] = *(const bf16x8*)(bs + n * BSTR + kb + 16);
    return u.v;
}

// ---- B tile staging (global -> regs -> LDS n-major transposed) --------------
template <bool BTRANS, bool NGUARD>
__device__ inline void stage_b_load(const __bf16* __restrict__ B, int N, int K,
                                    int bn, int kt, int t, bf16x8 r[2]) {
    if constexpr (!BTRANS) {
        // B is [K][N]; thread reads 8 cols of two adjacent K rows (k2, k2+1).
        int k2 = (t >> 4) * 2;
        int nb = (t & 15) * 8;
        int gn = bn * BN + nb;
        if (!NGUARD || gn + 7 < N) {
            const __bf16* p = B + (size_t)(kt * BK + k2) * N + gn;
            r[0] = *(const bf16x8*)p;
            r[1] = *(const bf16x8*)(p + N);
        } else { r[0] = bzero8(); r[1] = bzero8(); }
    } else {
        // B is [N][K] (weight.T form); thread reads 16 contiguous K of one row n.
        int n  = t >> 1;
        int kh = (t & 1) * 16;
        int gn = bn * BN + n;
        if (!NGUARD || gn < N) {
            const __bf16* p = B + (size_t)gn * K + kt * BK + kh;
            r[0] = *(const bf16x8*)p;
            r[1] = *(const bf16x8*)(p + 8);
        } else { r[0] = bzero8(); r[1] = bzero8(); }
    }
}

template <bool BTRANS>
__device__ inline void stage_b_store(__bf16* bs, int t, const bf16x8 r[2]) {
    if constexpr (!BTRANS) {
        int k2 = (t >> 4) * 2;
        int nb = (t & 15) * 8;
#pragma unroll
        for (int p = 0; p < 8; ++p) {
            union { unsigned int u; __bf16 b[2]; } pk;
            pk.b[0] = r[0][p];
            pk.b[1] = r[1][p];
            *(unsigned int*)(bs + (size_t)(nb + p) * BSTR + k2) = pk.u; // ds_store_b32
        }
    } else {
        int n  = t >> 1;
        int kh = (t & 1) * 16;
        *(bf16x8*)(bs + (size_t)n * BSTR + kh)     = r[0];
        *(bf16x8*)(bs + (size_t)n * BSTR + kh + 8) = r[1];
    }
}

// ---- A tile staging ---------------------------------------------------------
__device__ inline void stage_a_load(const __bf16* __restrict__ A, int K, int bm,
                                    int kt, int t, bf16x8 r[2]) {
    int rr = t >> 1;
    int c  = (t & 1) * 16;
    const __bf16* p = A + (size_t)(bm * BM + rr) * K + kt * BK + c;
    r[0] = *(const bf16x8*)p;
    r[1] = *(const bf16x8*)(p + 8);
}

__device__ inline void stage_a_store(__bf16* as, int t, const bf16x8 r[2]) {
    int rr = t >> 1;
    int c  = (t & 1) * 16;
    *(bf16x8*)(as + rr * BK + c)     = r[0];
    *(bf16x8*)(as + rr * BK + c + 8) = r[1];
}

#if USE_TDM
// Build a D# (ISA 8.3-8.5) for a 2D 128-row x 16-dword tile and fire the TDM.
// data_size=4B; tensor_dim0_stride = K/2 dwords; groups 2/3 zero (2D tensor).
// This toolchain's builtin takes 6 args: (g0, g1, g2, g3, g_extra, cpol).
__device__ inline void tdm_load_a(const __bf16* A, int K, int bm, int kt, __bf16* as) {
    unsigned long long g =
        (unsigned long long)(size_t)(A + (size_t)bm * BM * K + (size_t)kt * BK);
    unsigned int lds = (unsigned int)(size_t)as;   // low 32 bits = LDS offset
    u32x4 g0;
    g0[0] = 1u;                                     // count=1, user descriptor
    g0[1] = lds;                                    // lds_addr
    g0[2] = (unsigned int)g;                        // global_addr[31:0]
    g0[3] = ((unsigned int)(g >> 32) & 0x01FFFFFFu) // global_addr[56:32]
          | 0x80000000u;                            // type=2 ("image")
    unsigned int td0 = (unsigned int)(K >> 1);      // tensor_dim0 (dwords)
    unsigned int td1 = (unsigned int)NPTS;          // tensor_dim1 (rows)
    unsigned long long s0 = (unsigned long long)(K >> 1); // row stride (dwords)
    i32x8 g1;
    g1[0] = (int)(2u << 16);                        // data_size=4B, mask=0
    g1[1] = (int)((td0 & 0xFFFFu) << 16);           // barrier_addr=0 | td0 lo
    g1[2] = (int)((td0 >> 16) | ((td1 & 0xFFFFu) << 16));
    g1[3] = (int)((td1 >> 16) | (16u << 16));       // tile_dim0 = 16 dwords
    g1[4] = (int)128u;                              // tile_dim1 = 128, tile_dim2 = 0
    g1[5] = (int)(unsigned int)(s0 & 0xFFFFFFFFull);
    g1[6] = (int)(unsigned int)((s0 >> 32) & 0xFFFFull);
    g1[7] = 0;
    i32x4 z; z[0] = 0; z[1] = 0; z[2] = 0; z[3] = 0;
    i32x8 z8; z8[0] = 0; z8[1] = 0; z8[2] = 0; z8[3] = 0;
    z8[4] = 0; z8[5] = 0; z8[6] = 0; z8[7] = 0;
    __builtin_amdgcn_tensor_load_to_lds(g0, g1, z, z, z8, 0);
}
#endif

// ---------------------------------------------------------------------------
// C[M,N] = A[M,K](bf16,row-major) * B(bf16)   (B: [K][N], or [N][K] if BTRANS)
// Fused epilogue: per-row scale, bias, leaky-relu, f32 store, (+-)bf16 store,
// and dot(C, H) accumulation for the energy term.
// ---------------------------------------------------------------------------
template <bool BTRANS, bool NGUARD>
__global__ __launch_bounds__(256) void gemm_bf16_wmma(
    const __bf16* __restrict__ A, const __bf16* __restrict__ B,
    int M, int N, int K,
    const float* __restrict__ rowscale,
    const float* __restrict__ bias, int leaky,
    float* __restrict__ outF, __bf16* __restrict__ outB, int negB,
    float* __restrict__ dotP, const float* __restrict__ dotH)
{
    __shared__ __bf16 As[2][BM * BK];
    __shared__ __bf16 Bs[2][BN * BSTR];

    const int t    = threadIdx.x;
    const int lane = t & 31;
    const int wave = t >> 5;
    const int bm   = blockIdx.y;
    const int bn   = blockIdx.x;
    const int KT   = K / BK;

    const int wr = (wave >> 1) * 32;   // wave row base inside tile (4 waves in M)
    const int wc = (wave & 1) * 64;    // wave col base inside tile (2 waves in N)

    f32x8 acc[2][4];
#pragma unroll
    for (int i = 0; i < 2; ++i)
#pragma unroll
        for (int j = 0; j < 4; ++j)
#pragma unroll
            for (int g = 0; g < 8; ++g) acc[i][j][g] = 0.0f;

    bf16x8 breg[2];
#if !USE_TDM
    bf16x8 areg[2];
#endif

    // prologue: stage K-tile 0 into buffer 0
    stage_b_load<BTRANS, NGUARD>(B, N, K, bn, 0, t, breg);
#if USE_TDM
    if (wave == 0) tdm_load_a(A, K, bm, 0, &As[0][0]);
#else
    stage_a_load(A, K, bm, 0, t, areg);
    stage_a_store(&As[0][0], t, areg);
#endif
    stage_b_store<BTRANS>(&Bs[0][0], t, breg);

    // steady state: always have a next tile to stage (last iteration peeled)
    int buf = 0;
    for (int kt = 0; kt < KT - 1; ++kt) {
#if USE_TDM
        if (wave == 0) __builtin_amdgcn_s_wait_tensorcnt(0);
#endif
        __syncthreads();   // tile `buf` (TDM A + DS B) visible to all waves

        stage_b_load<BTRANS, NGUARD>(B, N, K, bn, kt + 1, t, breg);
#if USE_TDM
        if (wave == 0) tdm_load_a(A, K, bm, kt + 1, &As[buf ^ 1][0]);
#else
        stage_a_load(A, K, bm, kt + 1, t, areg);
#endif

        {   // compute on buffer `buf`
            const __bf16* as = &As[buf][0];
            const __bf16* bs = &Bs[buf][0];
            bf16x16 af[2], bfv[4];
#pragma unroll
            for (int i = 0; i < 2; ++i) af[i] = load_frag_a(as, wr + i * 16, lane);
#pragma unroll
            for (int j = 0; j < 4; ++j) bfv[j] = load_frag_b(bs, wc + j * 16, lane);
#pragma unroll
            for (int i = 0; i < 2; ++i)
#pragma unroll
                for (int j = 0; j < 4; ++j)
                    acc[i][j] = __builtin_amdgcn_wmma_f32_16x16x32_bf16(
                        false, af[i], false, bfv[j], (short)0, acc[i][j], false, false);
        }

#if !USE_TDM
        stage_a_store(&As[buf ^ 1][0], t, areg);
#endif
        stage_b_store<BTRANS>(&Bs[buf ^ 1][0], t, breg);
        __syncthreads();
        buf ^= 1;
    }

    // peeled final tile: compute only
#if USE_TDM
    if (wave == 0) __builtin_amdgcn_s_wait_tensorcnt(0);
#endif
    __syncthreads();
    {
        const __bf16* as = &As[buf][0];
        const __bf16* bs = &Bs[buf][0];
        bf16x16 af[2], bfv[4];
#pragma unroll
        for (int i = 0; i < 2; ++i) af[i] = load_frag_a(as, wr + i * 16, lane);
#pragma unroll
        for (int j = 0; j < 4; ++j) bfv[j] = load_frag_b(bs, wc + j * 16, lane);
#pragma unroll
        for (int i = 0; i < 2; ++i)
#pragma unroll
            for (int j = 0; j < 4; ++j)
                acc[i][j] = __builtin_amdgcn_wmma_f32_16x16x32_bf16(
                    false, af[i], false, bfv[j], (short)0, acc[i][j], false, false);
    }

    // ---- epilogue (C layout ISA 7.12.2: vgpr g -> M=g+8*(lane>=16), lane -> N) ----
    float dotLocal = 0.0f;
#pragma unroll
    for (int i = 0; i < 2; ++i) {
#pragma unroll
        for (int j = 0; j < 4; ++j) {
            int m0 = bm * BM + wr + i * 16 + ((lane >> 4) * 8);
            int n  = bn * BN + wc + j * 16 + (lane & 15);
            bool nok = (!NGUARD) || (n < N);
#pragma unroll
            for (int g = 0; g < 8; ++g) {
                int   m = m0 + g;
                float x = acc[i][j][g];
                if (rowscale) x *= rowscale[m];
                if (nok) {
                    if (bias)  x += bias[n];
                    if (leaky) x = (x > 0.0f) ? x : NEG_SLOPE * x;
                    size_t idx = (size_t)m * N + n;
                    if (outF) outF[idx] = x;
                    if (outB) outB[idx] = (__bf16)(negB ? -x : x);
                    if (dotP) dotLocal += x * dotH[idx];
                }
            }
        }
    }
    if (dotP) atomicAdd(dotP, dotLocal);
}

// ---------------------------------------------------------------------------
// small helper kernels
// ---------------------------------------------------------------------------
__global__ void cvt_f32_to_bf16(const float* __restrict__ s, __bf16* __restrict__ d, long n) {
    long i0 = ((long)blockIdx.x * blockDim.x + threadIdx.x) * 4;
    long st = (long)gridDim.x * blockDim.x * 4;
    for (long k = i0; k < n; k += st) {
        float4 v = *(const float4*)(s + k);
        union { __bf16 b[4]; unsigned long long q; } pk;
        pk.b[0] = (__bf16)v.x; pk.b[1] = (__bf16)v.y;
        pk.b[2] = (__bf16)v.z; pk.b[3] = (__bf16)v.w;
        *(unsigned long long*)(d + k) = pk.q;
    }
}

__global__ void spectrum_kernel(const float* __restrict__ S, const float* shift,
                                const float* expo, float* __restrict__ Sp, int n) {
    int i = blockIdx.x * blockDim.x + threadIdx.x;
    if (i < n) Sp[i] = __powf(shift[0] + S[i], expo[0]);
}

__global__ void normsq_kernel(const float* __restrict__ h, long n, float* out) {
    long i0 = (long)blockIdx.x * blockDim.x + threadIdx.x;
    long st = (long)gridDim.x * blockDim.x;
    float s = 0.0f;
    for (long k = i0; k < n; k += st) { float v = h[k]; s += v * v; }
    atomicAdd(out, s);
}

// Heun update: h <- h - ss*L0 - 0.5*ss^2*L1 ; emit bf16 copy + ||h||^2 partial
__global__ void update_h_kernel(float* __restrict__ h, const float* __restrict__ L0,
                                const float* __restrict__ L1, const float* ssp,
                                __bf16* __restrict__ hbf, float* normsq, long n) {
    float ss = ssp[0];
    float c2 = 0.5f * ss * ss;
    long i0 = (long)blockIdx.x * blockDim.x + threadIdx.x;
    long st = (long)gridDim.x * blockDim.x;
    float accn = 0.0f;
    for (long k = i0; k < n; k += st) {
        float hv = h[k] - ss * L0[k] - c2 * L1[k];
        h[k]   = hv;
        hbf[k] = (__bf16)hv;
        accn  += hv * hv;
    }
    atomicAdd(normsq, accn);
}

__global__ void finalize_energy(const float* __restrict__ stats, float* __restrict__ e, int n) {
    int i = threadIdx.x;
    if (i < n) e[i] = 0.5f * stats[i * 2 + 1] / stats[i * 2 + 0];
}

// ---------------------------------------------------------------------------
extern "C" void kernel_launch(void* const* d_in, const int* in_sizes, int n_in,
                              void* d_out, int out_size, void* d_ws, size_t ws_size,
                              hipStream_t stream)
{
    (void)in_sizes; (void)n_in; (void)out_size; (void)ws_size;
    const float* x     = (const float*)d_in[0];
    const float* U     = (const float*)d_in[1];
    const float* S     = (const float*)d_in[2];
    const float* Vh    = (const float*)d_in[3];
    const float* snl   = (const float*)d_in[4];
    const float* enc_w = (const float*)d_in[5];
    const float* enc_b = (const float*)d_in[6];
    const float* W     = (const float*)d_in[7];
    const float* dw0   = (const float*)d_in[8];
    const float* db0   = (const float*)d_in[9];
    const float* dw1   = (const float*)d_in[10];
    const float* db1   = (const float*)d_in[11];
    const float* shift = (const float*)d_in[12];
    const float* expo  = (const float*)d_in[13];
    const float* ssp   = (const float*)d_in[14];

    char* ws = (char*)d_ws;
    size_t off = 0;
    auto alloc = [&](size_t bytes) {
        char* p = ws + off;
        off = (off + bytes + 255) & ~(size_t)255;
        return (void*)p;
    };

    const size_t NN2 = (size_t)NPTS * NPTS;
    const size_t NH  = (size_t)NPTS * HID;

    __bf16* Ubf   = (__bf16*)alloc(NN2 * 2);
    __bf16* Vhbf  = (__bf16*)alloc(NN2 * 2);
    __bf16* snlbf = (__bf16*)alloc(NN2 * 2);
    __bf16* Wbf   = (__bf16*)alloc((size_t)HID * HID * 2);
    __bf16* xbf   = (__bf16*)alloc((size_t)NPTS * IN_C * 2);
    __bf16* ewbf  = (__bf16*)alloc((size_t)HID * IN_C * 2);
    __bf16* d0bf  = (__bf16*)alloc((size_t)HID * HID * 2);
    __bf16* d1bf  = (__bf16*)alloc((size_t)OUT_C * HID * 2);
    float*  Sp    = (float*) alloc((size_t)NPTS * 4);
    float*  h     = (float*) alloc(NH * 4);
    __bf16* hbf   = (__bf16*)alloc(NH * 2);
    __bf16* T1bf  = (__bf16*)alloc(NH * 2);
    __bf16* T2bf  = (__bf16*)alloc(NH * 2);
    __bf16* K0bf  = (__bf16*)alloc(NH * 2);
    __bf16* dbf   = (__bf16*)alloc(NH * 2);
    float*  L0    = (float*) alloc(NH * 4);
    float*  L1    = (float*) alloc(NH * 4);
    float*  stats = (float*) alloc((NUM_LAYERS + 1) * 2 * 4);

    (void)hipMemsetAsync(stats, 0, (NUM_LAYERS + 1) * 2 * sizeof(float), stream);

    auto cvt = [&](const float* s, __bf16* d, long n) {
        int blocks = (int)((n / 4 + 255) / 256);
        if (blocks > 4096) blocks = 4096;
        cvt_f32_to_bf16<<<blocks, 256, 0, stream>>>(s, d, n);
    };
    cvt(U,     Ubf,   (long)NN2);
    cvt(Vh,    Vhbf,  (long)NN2);
    cvt(snl,   snlbf, (long)NN2);
    cvt(W,     Wbf,   (long)HID * HID);
    cvt(x,     xbf,   (long)NPTS * IN_C);
    cvt(enc_w, ewbf,  (long)HID * IN_C);
    cvt(dw0,   d0bf,  (long)HID * HID);
    cvt(dw1,   d1bf,  (long)OUT_C * HID);

    spectrum_kernel<<<NPTS / 256, 256, 0, stream>>>(S, shift, expo, Sp, NPTS);

    // bTrans / N-guard dispatch (guard only needed when N is not a tile multiple)
    auto gemm = [&](const __bf16* A, const __bf16* B, int M, int N, int K, int bTrans,
                    const float* rs, const float* bias, int leaky,
                    float* oF, __bf16* oB, int neg, float* dp, const float* dh) {
        dim3 grid((N + BN - 1) / BN, M / BM);
        if (!bTrans) {
            if (N % BN == 0)
                gemm_bf16_wmma<false, false><<<grid, 256, 0, stream>>>(
                    A, B, M, N, K, rs, bias, leaky, oF, oB, neg, dp, dh);
            else
                gemm_bf16_wmma<false, true><<<grid, 256, 0, stream>>>(
                    A, B, M, N, K, rs, bias, leaky, oF, oB, neg, dp, dh);
        } else {
            if (N % BN == 0)
                gemm_bf16_wmma<true, false><<<grid, 256, 0, stream>>>(
                    A, B, M, N, K, rs, bias, leaky, oF, oB, neg, dp, dh);
            else
                gemm_bf16_wmma<true, true><<<grid, 256, 0, stream>>>(
                    A, B, M, N, K, rs, bias, leaky, oF, oB, neg, dp, dh);
        }
    };

    // encoder: h = x @ enc_w.T + enc_b   (enc_w stored [N][K] -> bTrans path)
    gemm(xbf, ewbf, NPTS, HID, IN_C, 1, nullptr, enc_b, 0, h, hbf, 0, nullptr, nullptr);

    // energy(h0): ||h||^2 and h . (snl h)
    normsq_kernel<<<512, 256, 0, stream>>>(h, (long)NH, &stats[0]);
    gemm(snlbf, hbf, NPTS, HID, NPTS, 0, nullptr, nullptr, 0,
         nullptr, nullptr, 0, &stats[1], h);

    for (int l = 0; l < NUM_LAYERS; ++l) {
        // L0 = U @ (Sp * (Vh h)) @ W ; K0bf = bf16(-L0)
        gemm(Vhbf, hbf,  NPTS, HID, NPTS, 0, Sp, nullptr, 0, nullptr, T1bf, 0, nullptr, nullptr);
        gemm(T1bf, Wbf,  NPTS, HID, HID,  0, nullptr, nullptr, 0, nullptr, T2bf, 0, nullptr, nullptr);
        gemm(Ubf,  T2bf, NPTS, HID, NPTS, 0, nullptr, nullptr, 0, L0, K0bf, 1, nullptr, nullptr);
        // L1 = lxw(K0)
        gemm(Vhbf, K0bf, NPTS, HID, NPTS, 0, Sp, nullptr, 0, nullptr, T1bf, 0, nullptr, nullptr);
        gemm(T1bf, Wbf,  NPTS, HID, HID,  0, nullptr, nullptr, 0, nullptr, T2bf, 0, nullptr, nullptr);
        gemm(Ubf,  T2bf, NPTS, HID, NPTS, 0, nullptr, nullptr, 0, L1, nullptr, 0, nullptr, nullptr);
        // h <- h - ss*L0 - 0.5 ss^2 L1 ; refresh hbf ; accumulate ||h||^2
        update_h_kernel<<<1024, 256, 0, stream>>>(h, L0, L1, ssp, hbf,
                                                  &stats[(l + 1) * 2 + 0], (long)NH);
        // energy dot
        gemm(snlbf, hbf, NPTS, HID, NPTS, 0, nullptr, nullptr, 0,
             nullptr, nullptr, 0, &stats[(l + 1) * 2 + 1], h);
    }

    // decoder
    gemm(hbf, d0bf, NPTS, HID,   HID, 1, nullptr, db0, 1, nullptr, dbf, 0, nullptr, nullptr);
    gemm(dbf, d1bf, NPTS, OUT_C, HID, 1, nullptr, db1, 0, (float*)d_out, nullptr, 0, nullptr, nullptr);

    finalize_energy<<<1, 32, 0, stream>>>(stats,
        (float*)d_out + (size_t)NPTS * OUT_C, NUM_LAYERS + 1);
}
